// PhysicsAttention_58858231824691
// MI455X (gfx1250) — matrix operand order repaired and checked by reading, not compiled, not run on hardware
//
#include <hip/hip_runtime.h>
#include <hip/hip_bf16.h>
#include <stdint.h>

#define B_  8
#define N_  16384
#define C_  512
#define H_  8
#define D_  64
#define G_  64
#define M_  (B_*N_)   /* 131072 rows */

typedef __attribute__((ext_vector_type(16))) __bf16 v16bf;
typedef __attribute__((ext_vector_type(8)))  float  v8f;
typedef __attribute__((ext_vector_type(4)))  unsigned int v4u;
typedef __attribute__((ext_vector_type(8)))  int v8i_;
typedef __attribute__((ext_vector_type(4)))  int v4i_;

// ---------- bf16 <-> f32 (bit-level, round-to-nearest-even) ----------
static __device__ __forceinline__ __bf16 f2bf(float f) {
  union { float f; uint32_t u; } a; a.f = f;
  uint32_t r = (a.u + 0x7fffu + ((a.u >> 16) & 1u)) >> 16;
  union { unsigned short s; __bf16 b; } o; o.s = (unsigned short)r;
  return o.b;
}
static __device__ __forceinline__ float bf2f(__bf16 b) {
  union { __bf16 b; unsigned short s; } i; i.b = b;
  union { uint32_t u; float f; } o; o.u = ((uint32_t)i.s) << 16;
  return o.f;
}

// ---------- WMMA fragment loaders from LDS ----------
// A (16x32 bf16): lane m = m0+(lane&15), kbase = 8*(lane>>4);
// halves 0..7 -> k kbase..kbase+7, halves 8..15 -> k kbase+16..kbase+23.
static __device__ __forceinline__ v16bf load_a_frag(const __bf16* s, int ld, int m0, int k0, int lane) {
  int m  = m0 + (lane & 15);
  int kb = k0 + ((lane >> 4) << 3);
  const __bf16* p = s + m * ld + kb;
  v16bf a;
  ((float4*)&a)[0] = *(const float4*)(p);
  ((float4*)&a)[1] = *(const float4*)(p + 16);
  return a;
}
// B (32x16 bf16) staged as [n][k]: lane n = n0+(lane&15), kbase = 8*(lane>>4);
// halves 0..7 -> k kbase..kbase+7, halves 8..15 -> k kbase+16..kbase+23.
static __device__ __forceinline__ v16bf load_b_frag(const __bf16* s, int ld, int n0, int k0, int lane) {
  int n  = n0 + (lane & 15);
  int kb = k0 + ((lane >> 4) << 3);
  const __bf16* p = s + n * ld + kb;
  v16bf b;
  ((float4*)&b)[0] = *(const float4*)(p);
  ((float4*)&b)[1] = *(const float4*)(p + 16);
  return b;
}
static __device__ __forceinline__ v8f wmma_bf16(v16bf a, v16bf b, v8f c) {
  return __builtin_amdgcn_wmma_f32_16x16x32_bf16(false, a, false, b, (short)0, c, false, false);
}
#define V8F_ZERO {0.f,0.f,0.f,0.f,0.f,0.f,0.f,0.f}

// ---------- Tensor Data Mover: 2-D bf16 tile load, global -> LDS ----------
// D# per CDNA5 ISA (2-D: groups 2/3 zero).
// This toolchain exposes the 6-arg builtin: (v4u, v8i, v4i, v4i, v8i, i32 cpol).
static __device__ __forceinline__ unsigned lds_off(const void* p) {
  return (unsigned)(unsigned long long)(uintptr_t)p;   // low 32 bits = LDS offset
}
static __device__ __forceinline__ void tdm_load_2d(
    unsigned lds_addr, const void* gaddr,
    unsigned tensor_d0, unsigned tensor_d1, unsigned d0_stride,
    unsigned tile_d0, unsigned tile_d1) {
  unsigned long long ga = (unsigned long long)(uintptr_t)gaddr;
  v4u g0;
  g0[0] = 1u;                                           // count=1 (valid), user mode
  g0[1] = lds_addr;                                     // lds_addr [63:32]
  g0[2] = (unsigned)(ga & 0xffffffffu);                 // global_addr low
  g0[3] = (unsigned)((ga >> 32) & 0x01ffffffu) | (2u << 30);  // addr hi | type=2
  v8i_ g1;
  g1[0] = (1 << 16);                                    // data_size=1 (2B), wg_mask=0
  g1[1] = (int)((tensor_d0 & 0xffffu) << 16);           // tensor_dim0[15:0] @ [63:48]
  g1[2] = (int)(((tensor_d0 >> 16) & 0xffffu) | ((tensor_d1 & 0xffffu) << 16));
  g1[3] = (int)(((tensor_d1 >> 16) & 0xffffu) | ((tile_d0 & 0xffffu) << 16));
  g1[4] = (int)(tile_d1 & 0xffffu);                     // tile_dim1; tile_dim2=0
  g1[5] = (int)d0_stride;                               // tensor_dim0_stride[31:0]
  g1[6] = 0;
  g1[7] = 0;
  v4i_ zz4 = {0, 0, 0, 0};
  v8i_ zz8 = {0, 0, 0, 0, 0, 0, 0, 0};
  __builtin_amdgcn_tensor_load_to_lds(g0, g1, zz4, zz4, zz8, 0);
}

// ---------- elementwise helpers ----------
__global__ void cvt_kernel(const float* __restrict__ in, __bf16* __restrict__ out, int n) {
  int i = blockIdx.x * 256 + threadIdx.x;
  if (i < n) out[i] = f2bf(in[i]);
}
// convert + transpose: out[c][r] = in[r][c]
__global__ void cvtT_kernel(const float* __restrict__ in, __bf16* __restrict__ out,
                            int rows, int cols) {
  int i = blockIdx.x * 256 + threadIdx.x;
  if (i < rows * cols) {
    int r = i / cols, c = i % cols;
    out[(size_t)c * rows + r] = f2bf(in[i]);
  }
}
__global__ void zero_kernel(float* __restrict__ p, int n) {
  int i = blockIdx.x * 256 + threadIdx.x;
  if (i < n) p[i] = 0.f;
}

// ---------- main GEMM: C[Mx512] = A[Mx512](bf16) @ W (Wt transposed [n][k]) + bias ----------
// TDM double-buffered K slabs of 128; block tile 128x64, wave tile 32x32 (4 WMMA/chunk).
// mode 0: bf16 row-major; mode 1: bf16 into fxT (b,h,d,n); mode 2: f32 row-major.
__global__ __launch_bounds__(256) void wmma_gemm512_tdm(
    const __bf16* __restrict__ A, const __bf16* __restrict__ Wt,
    const float* __restrict__ bias, void* __restrict__ out, int mode) {
  __shared__ __align__(16) __bf16 sA[2][128 * 128];   // 2 x 32 KB
  __shared__ __align__(16) __bf16 sB[2][64 * 128];    // 2 x 16 KB
  const int t = threadIdx.x, lane = t & 31, wave = t >> 5;
  const int row0 = blockIdx.y * 128, col0 = blockIdx.x * 64;
  const int m0 = (wave >> 1) * 32, n0 = (wave & 1) * 32;
  v8f acc00 = V8F_ZERO, acc01 = V8F_ZERO, acc10 = V8F_ZERO, acc11 = V8F_ZERO;
  const bool issuer = (wave == 0);
  if (issuer) {
    tdm_load_2d(lds_off(sA[0]), A  + (size_t)row0 * 512, 512, M_,  512, 128, 128);
    tdm_load_2d(lds_off(sB[0]), Wt + (size_t)col0 * 512, 512, 512, 512, 128, 64);
  }
  for (int c = 0; c < 4; ++c) {
    if (issuer) {
      if (c + 1 < 4) {
        tdm_load_2d(lds_off(sA[(c + 1) & 1]), A  + (size_t)row0 * 512 + (c + 1) * 128,
                    512, M_, 512, 128, 128);
        tdm_load_2d(lds_off(sB[(c + 1) & 1]), Wt + (size_t)col0 * 512 + (c + 1) * 128,
                    512, 512, 512, 128, 64);
        __builtin_amdgcn_s_wait_tensorcnt((short)2);   // current slab complete
      } else {
        __builtin_amdgcn_s_wait_tensorcnt((short)0);
      }
    }
    __syncthreads();
    const __bf16* pa = sA[c & 1];
    const __bf16* pb = sB[c & 1];
    #pragma unroll
    for (int ks = 0; ks < 128; ks += 32) {
      v16bf a0 = load_a_frag(pa, 128, m0,      ks, lane);
      v16bf a1 = load_a_frag(pa, 128, m0 + 16, ks, lane);
      v16bf b0 = load_b_frag(pb, 128, n0,      ks, lane);
      v16bf b1 = load_b_frag(pb, 128, n0 + 16, ks, lane);
      acc00 = wmma_bf16(a0, b0, acc00);
      acc01 = wmma_bf16(a0, b1, acc01);
      acc10 = wmma_bf16(a1, b0, acc10);
      acc11 = wmma_bf16(a1, b1, acc11);
    }
    __syncthreads();
  }
  const int lm = (lane >> 4) << 3, nn = lane & 15;
  const int cA = col0 + n0 + nn, cB = cA + 16;
  const float bb0 = bias[cA], bb1 = bias[cB];
  #pragma unroll
  for (int v = 0; v < 8; ++v) {
    int rA = row0 + m0 + v + lm, rB = rA + 16;
    float v00 = acc00[v] + bb0, v01 = acc01[v] + bb1;
    float v10 = acc10[v] + bb0, v11 = acc11[v] + bb1;
    if (mode == 0) {
      __bf16* o = (__bf16*)out;
      o[(size_t)rA * 512 + cA] = f2bf(v00); o[(size_t)rA * 512 + cB] = f2bf(v01);
      o[(size_t)rB * 512 + cA] = f2bf(v10); o[(size_t)rB * 512 + cB] = f2bf(v11);
    } else if (mode == 2) {
      float* o = (float*)out;
      o[(size_t)rA * 512 + cA] = v00; o[(size_t)rA * 512 + cB] = v01;
      o[(size_t)rB * 512 + cA] = v10; o[(size_t)rB * 512 + cB] = v11;
    } else {  // mode 1: fxT[(b*H+h)*64 + d][n], b=r>>14, n=r&16383, h=c>>6, d=c&63
      __bf16* o = (__bf16*)out;
      int h = cA >> 6;
      size_t zA = ((size_t)((rA >> 14) * H_ + h) * 64);
      size_t zB = ((size_t)((rB >> 14) * H_ + h) * 64);
      o[(zA + (cA & 63)) * N_ + (rA & (N_ - 1))] = f2bf(v00);
      o[(zA + (cB & 63)) * N_ + (rA & (N_ - 1))] = f2bf(v01);
      o[(zB + (cA & 63)) * N_ + (rB & (N_ - 1))] = f2bf(v10);
      o[(zB + (cB & 63)) * N_ + (rB & (N_ - 1))] = f2bf(v11);
    }
  }
}

// ---------- slice logits: per head h, logits = x_mid[:, h*64:+64] @ Wslice + bslice ----------
// WslT is Wslice transposed ([g][k]); output bf16 in sw (b,h,n,g). grid (M/64, H).
__global__ __launch_bounds__(256) void wmma_slice_logits(
    const __bf16* __restrict__ xmid, const __bf16* __restrict__ WslT,
    const float* __restrict__ bsl, __bf16* __restrict__ sw) {
  __shared__ __align__(16) __bf16 sA[64 * 32];
  __shared__ __align__(16) __bf16 sB[64 * 32];
  const int t = threadIdx.x, lane = t & 31, wave = t >> 5;
  const int row0 = blockIdx.x * 64, h = blockIdx.y;
  const int m0 = (wave >> 1) * 16, n0 = (wave & 1) * 32;
  v8f acc0 = V8F_ZERO, acc1 = V8F_ZERO;
  const int ar = t >> 2, ac = (t & 3) << 3;
  for (int kc = 0; kc < 64; kc += 32) {
    *(uint4*)(sA + ar * 32 + ac) =
        *(const uint4*)(xmid + (size_t)(row0 + ar) * 512 + h * 64 + kc + ac);
    *(uint4*)(sB + ar * 32 + ac) =
        *(const uint4*)(WslT + (size_t)ar * 64 + kc + ac);
    __syncthreads();
    v16bf af = load_a_frag(sA, 32, m0, 0, lane);
    v16bf b0 = load_b_frag(sB, 32, n0, 0, lane);
    v16bf b1 = load_b_frag(sB, 32, n0 + 16, 0, lane);
    acc0 = wmma_bf16(af, b0, acc0);
    acc1 = wmma_bf16(af, b1, acc1);
    __syncthreads();
  }
  const int lm = (lane >> 4) << 3, nn = lane & 15;
  const int g0 = n0 + nn, g1 = g0 + 16;
  const float bb0 = bsl[g0], bb1 = bsl[g1];
  #pragma unroll
  for (int v = 0; v < 8; ++v) {
    int row = row0 + m0 + v + lm;
    int b = row >> 14, n = row & (N_ - 1);
    size_t base = ((size_t)(b * H_ + h) * N_ + n) * 64;
    sw[base + g0] = f2bf(acc0[v] + bb0);
    sw[base + g1] = f2bf(acc1[v] + bb1);
  }
}

// ---------- softmax over G=64 in place + transposed copy swT + slice_norm ----------
__global__ __launch_bounds__(256) void softmax_kernel(
    __bf16* __restrict__ sw, __bf16* __restrict__ swT,
    const float* __restrict__ temperature, float* __restrict__ norm) {
  __shared__ float pn[64];
  const int t = threadIdx.x, lane = t & 31, wave = t >> 5;
  const int R = blockIdx.x * 8 + wave;
  const int h = (R >> 14) & (H_ - 1);
  if (t < 64) pn[t] = 0.f;
  __syncthreads();
  float tp = fminf(fmaxf(temperature[h], 0.1f), 5.0f);
  const size_t base = (size_t)R * 64;
  float w0 = bf2f(sw[base + lane]) / tp;
  float w1 = bf2f(sw[base + 32 + lane]) / tp;
  float mx = fmaxf(w0, w1);
  #pragma unroll
  for (int o = 16; o >= 1; o >>= 1) mx = fmaxf(mx, __shfl_xor(mx, o, 32));
  float e0 = __expf(w0 - mx), e1 = __expf(w1 - mx);
  float s = e0 + e1;
  #pragma unroll
  for (int o = 16; o >= 1; o >>= 1) s += __shfl_xor(s, o, 32);
  float inv = 1.f / s;
  float r0 = e0 * inv, r1 = e1 * inv;
  sw[base + lane]      = f2bf(r0);
  sw[base + 32 + lane] = f2bf(r1);
  const int zbh = R >> 14;             // b*H + h (constant per block)
  const int n = R & (N_ - 1);
  swT[((size_t)zbh * 64 + lane)      * N_ + n] = f2bf(r0);
  swT[((size_t)zbh * 64 + lane + 32) * N_ + n] = f2bf(r1);
  atomicAdd(&pn[lane], r0);
  atomicAdd(&pn[lane + 32], r1);
  __syncthreads();
  if (t < 64) atomicAdd(&norm[zbh * 64 + t], pn[t]);
}

// ---------- slice_token partial: stacc[z] += Sw[z]^T (GxN) @ fx[z] (NxD) ----------
// swT is (b,h,g,n); fxT is (b,h,d,n): both k(=n)-contiguous -> b128 staging.
__global__ __launch_bounds__(256) void wmma_slice_token(
    const __bf16* __restrict__ swT, const __bf16* __restrict__ fxT,
    float* __restrict__ stacc) {
  __shared__ __align__(16) __bf16 sA[64 * 32];   // [g][n]
  __shared__ __align__(16) __bf16 sB[64 * 32];   // [d][n]
  const int t = threadIdx.x, lane = t & 31, wave = t >> 5;
  const int z = blockIdx.x;
  const int m0 = (wave >> 1) * 16, n0 = (wave & 1) * 32;
  const int kstart = blockIdx.y * (N_ / 8);
  v8f acc0 = V8F_ZERO, acc1 = V8F_ZERO;
  const int ar = t >> 2, ac = (t & 3) << 3;
  for (int kc = kstart; kc < kstart + N_ / 8; kc += 32) {
    *(uint4*)(sA + ar * 32 + ac) =
        *(const uint4*)(swT + ((size_t)z * 64 + ar) * N_ + kc + ac);
    *(uint4*)(sB + ar * 32 + ac) =
        *(const uint4*)(fxT + ((size_t)z * 64 + ar) * N_ + kc + ac);
    __syncthreads();
    v16bf af = load_a_frag(sA, 32, m0, 0, lane);
    v16bf b0 = load_b_frag(sB, 32, n0, 0, lane);
    v16bf b1 = load_b_frag(sB, 32, n0 + 16, 0, lane);
    acc0 = wmma_bf16(af, b0, acc0);
    acc1 = wmma_bf16(af, b1, acc1);
    __syncthreads();
  }
  const int lm = (lane >> 4) << 3, nn = lane & 15;
  #pragma unroll
  for (int v = 0; v < 8; ++v) {
    int g = m0 + v + lm;
    atomicAdd(&stacc[(size_t)z * 4096 + g * 64 + n0 + nn], acc0[v]);
    atomicAdd(&stacc[(size_t)z * 4096 + g * 64 + n0 + nn + 16], acc1[v]);
  }
}

__global__ void st_norm_kernel(float* __restrict__ stacc, const float* __restrict__ norm) {
  int i = blockIdx.x * 256 + threadIdx.x;
  if (i < B_ * H_ * G_ * D_) {
    int z = i >> 12, g = (i >> 6) & 63;
    stacc[i] = stacc[i] / (norm[z * 64 + g] + 1e-5f);
  }
}

// ---------- tiny 64x64 attention per (b,h); writes out_slice TRANSPOSED [d][g] ----------
__global__ __launch_bounds__(256) void attn_kernel(
    const float* __restrict__ st, const float* __restrict__ Wq,
    const float* __restrict__ Wk, const float* __restrict__ Wv,
    __bf16* __restrict__ oslT) {
  __shared__ float s_st[4096];
  __shared__ float s_q[4096];
  __shared__ float s_k[4096];
  __shared__ float s_v[4096];
  const int z = blockIdx.x, t = threadIdx.x;
  for (int i = t; i < 4096; i += 256) s_st[i] = st[(size_t)z * 4096 + i];
  __syncthreads();
  for (int i = t; i < 4096; i += 256) {
    int g = i >> 6, d = i & 63;
    float aq = 0.f, ak = 0.f, av = 0.f;
    for (int c = 0; c < 64; ++c) {
      float s = s_st[g * 64 + c];
      aq = fmaf(s, Wq[c * 64 + d], aq);
      ak = fmaf(s, Wk[c * 64 + d], ak);
      av = fmaf(s, Wv[c * 64 + d], av);
    }
    s_q[i] = aq; s_k[i] = ak; s_v[i] = av;
  }
  __syncthreads();
  if (t < 64) {
    float mx = -3.0e38f;
    for (int j = 0; j < 64; ++j) {
      float dt = 0.f;
      for (int c = 0; c < 64; ++c) dt = fmaf(s_q[t * 64 + c], s_k[j * 64 + c], dt);
      dt *= 0.125f;                 // D^-0.5
      s_st[t * 64 + j] = dt;        // row t private to thread t
      mx = fmaxf(mx, dt);
    }
    float sum = 0.f;
    for (int j = 0; j < 64; ++j) {
      float e = __expf(s_st[t * 64 + j] - mx);
      s_st[t * 64 + j] = e; sum += e;
    }
    float inv = 1.f / sum;
    for (int d = 0; d < 64; ++d) {
      float o = 0.f;
      for (int j = 0; j < 64; ++j) o = fmaf(s_st[t * 64 + j], s_v[j * 64 + d], o);
      oslT[(size_t)z * 4096 + d * 64 + t] = f2bf(o * inv);   // transposed store
    }
  }
}

// ---------- deslice: out_x[z] (Nx64) = Sw[z] (NxG) @ out_slice[z] (GxD) ----------
// A = sw row-major (n,g); B from oslT [d][g] (k-contiguous). grid (N/64, B*H).
__global__ __launch_bounds__(256) void wmma_deslice(
    const __bf16* __restrict__ sw, const __bf16* __restrict__ oslT,
    __bf16* __restrict__ outx) {
  __shared__ __align__(16) __bf16 sA[64 * 32];
  __shared__ __align__(16) __bf16 sB[64 * 32];
  const int t = threadIdx.x, lane = t & 31, wave = t >> 5;
  const int row0 = blockIdx.x * 64, z = blockIdx.y, b = z >> 3, h = z & 7;
  const int m0 = (wave >> 1) * 16, n0 = (wave & 1) * 32;
  v8f acc0 = V8F_ZERO, acc1 = V8F_ZERO;
  const int ar = t >> 2, ac = (t & 3) << 3;
  for (int kc = 0; kc < 64; kc += 32) {
    *(uint4*)(sA + ar * 32 + ac) =
        *(const uint4*)(sw + ((size_t)z * N_ + row0 + ar) * 64 + kc + ac);
    *(uint4*)(sB + ar * 32 + ac) =
        *(const uint4*)(oslT + (size_t)z * 4096 + (size_t)ar * 64 + kc + ac);
    __syncthreads();
    v16bf af = load_a_frag(sA, 32, m0, 0, lane);
    v16bf b0 = load_b_frag(sB, 32, n0, 0, lane);
    v16bf b1 = load_b_frag(sB, 32, n0 + 16, 0, lane);
    acc0 = wmma_bf16(af, b0, acc0);
    acc1 = wmma_bf16(af, b1, acc1);
    __syncthreads();
  }
  const int lm = (lane >> 4) << 3, nn = lane & 15;
  const int d0 = n0 + nn, d1 = d0 + 16;
  #pragma unroll
  for (int v = 0; v < 8; ++v) {
    int n = row0 + m0 + v + lm;
    size_t base = ((size_t)b * N_ + n) * 512 + h * 64;
    outx[base + d0] = f2bf(acc0[v]);
    outx[base + d1] = f2bf(acc1[v]);
  }
}

// =====================================================================
extern "C" void kernel_launch(void* const* d_in, const int* in_sizes, int n_in,
                              void* d_out, int out_size, void* d_ws, size_t ws_size,
                              hipStream_t stream) {
  (void)in_sizes; (void)n_in; (void)out_size; (void)ws_size;
  const float* x           = (const float*)d_in[0];
  const float* temperature = (const float*)d_in[1];
  const float* Wx          = (const float*)d_in[2];
  const float* bx          = (const float*)d_in[3];
  const float* Wfx         = (const float*)d_in[4];
  const float* bfx         = (const float*)d_in[5];
  const float* Wslice      = (const float*)d_in[6];
  const float* bslice      = (const float*)d_in[7];
  const float* Wq          = (const float*)d_in[8];
  const float* Wk          = (const float*)d_in[9];
  const float* Wv          = (const float*)d_in[10];
  const float* Wout        = (const float*)d_in[11];
  const float* bout        = (const float*)d_in[12];
  float* out = (float*)d_out;

  const size_t S1 = (size_t)M_ * C_ * sizeof(__bf16);   // 134,217,728 bytes
  char* ws = (char*)d_ws;
  __bf16* xb    = (__bf16*)(ws);              // x bf16; reused as out_x later
  __bf16* xmid  = (__bf16*)(ws + S1);
  __bf16* fxT   = (__bf16*)(ws + 2 * S1);     // (b,h,d,n)
  __bf16* sw    = (__bf16*)(ws + 3 * S1);     // (b,h,n,g), softmaxed in place
  __bf16* swT   = (__bf16*)(ws + 4 * S1);     // (b,h,g,n)
  char* sm = ws + 5 * S1;
  __bf16* WxT   = (__bf16*)(sm);              // [n][k] transposed weights
  __bf16* WfxT  = WxT   + 512 * 512;
  __bf16* WoutT = WfxT  + 512 * 512;
  __bf16* WslT  = WoutT + 512 * 512;
  float*  norm  = (float*)(WslT + 64 * 64);   // 4096 floats
  float*  stacc = norm + 4096;                // 262144 floats
  __bf16* oslT  = (__bf16*)(stacc + 262144);  // out_slice transposed [d][g]
  __bf16* outx  = xb;

  { int n = 4096 + 262144; zero_kernel<<<(n + 255) / 256, 256, 0, stream>>>(norm, n); }
  { int n = M_ * C_;   cvt_kernel <<<(n + 255) / 256, 256, 0, stream>>>(x, xb, n); }
  { int n = 512 * 512; cvtT_kernel<<<(n + 255) / 256, 256, 0, stream>>>(Wx,   WxT,   512, 512); }
  { int n = 512 * 512; cvtT_kernel<<<(n + 255) / 256, 256, 0, stream>>>(Wfx,  WfxT,  512, 512); }
  { int n = 512 * 512; cvtT_kernel<<<(n + 255) / 256, 256, 0, stream>>>(Wout, WoutT, 512, 512); }
  { int n = 64 * 64;   cvtT_kernel<<<(n + 255) / 256, 256, 0, stream>>>(Wslice, WslT, 64, 64); }

  wmma_gemm512_tdm<<<dim3(8, M_ / 128), 256, 0, stream>>>(xb, WxT,  bx,  xmid, 0);
  wmma_gemm512_tdm<<<dim3(8, M_ / 128), 256, 0, stream>>>(xb, WfxT, bfx, fxT,  1);
  wmma_slice_logits<<<dim3(M_ / 64, H_), 256, 0, stream>>>(xmid, WslT, bslice, sw);
  softmax_kernel<<<(B_ * H_ * N_) / 8, 256, 0, stream>>>(sw, swT, temperature, norm);
  wmma_slice_token<<<dim3(B_ * H_, 8), 256, 0, stream>>>(swT, fxT, stacc);
  { int n = B_ * H_ * G_ * D_; st_norm_kernel<<<(n + 255) / 256, 256, 0, stream>>>(stacc, norm); }
  attn_kernel<<<B_ * H_, 256, 0, stream>>>(stacc, Wq, Wk, Wv, oslT);
  wmma_deslice<<<dim3(N_ / 64, B_ * H_), 256, 0, stream>>>(sw, oslT, outx);
  wmma_gemm512_tdm<<<dim3(8, M_ / 128), 256, 0, stream>>>(outx, WoutT, bout, out, 2);
}
